// Model_52810917871638
// MI455X (gfx1250) — matrix operand order repaired and checked by reading, not compiled, not run on hardware
//
#include <hip/hip_runtime.h>
#include <hip/hip_bf16.h>

typedef __attribute__((ext_vector_type(16))) _Float16 v16h;
typedef __attribute__((ext_vector_type(8)))  _Float16 v8h;
typedef __attribute__((ext_vector_type(8)))  float    v8f;

constexpr int NS = 256;        // sequence length
constexpr int NP = NS * NS;    // 65536 pair rows
constexpr int CH = 128;        // channels (C == CI)
constexpr int HD = 32;         // head dim
constexpr int NH = 4;          // heads

// ---------------- WMMA fragment loaders (CDNA5 16x16x32 f16 layouts) ----------------
__device__ __forceinline__ v16h cat8(v8h lo, v8h hi) {
  return __builtin_shufflevector(lo, hi, 0, 1, 2, 3, 4, 5, 6, 7, 8, 9, 10, 11, 12, 13, 14, 15);
}

// A (16x32 f16): lane L<16 = row L, K {0..7,16..23}; lane L+16 = row L, K {8..15,24..31}.
__device__ __forceinline__ v16h load_a_row(const _Float16* A, int lda) {
  int lane = threadIdx.x & 31;
  const _Float16* p = A + (size_t)(lane & 15) * lda + (lane >> 4) * 8;
  return cat8(*(const v8h*)p, *(const v8h*)(p + 16));
}

// B (32x16 f16) from a row-major [n][k] matrix (B = M^T): lane L<16 = col L, K {0..7,16..23}; lane L+16 = col L, K {8..15,24..31}.
__device__ __forceinline__ v16h load_b_rows(const _Float16* M, int ld) {
  int lane = threadIdx.x & 31;
  const _Float16* p = M + (size_t)(lane & 15) * ld + (lane >> 4) * 8;
  return cat8(*(const v8h*)p, *(const v8h*)(p + 16));
}

__device__ __forceinline__ v8f wmma32(v16h a, v16h b, v8f c) {
  return __builtin_amdgcn_wmma_f32_16x16x32_f16(false, a, false, b, (short)0, c, false, false);
}

// single 16-col tile
template <int KT>
__device__ __forceinline__ v8f gemm_tile_t(const _Float16* A, int lda,
                                           const _Float16* Bt, int ldbt) {
  v8f acc = {};
#pragma unroll
  for (int t = 0; t < KT; ++t) {
    v16h a = load_a_row(A + t * 32, lda);
    v16h b = load_b_rows(Bt + t * 32, ldbt);
    acc = wmma32(a, b, acc);
  }
  return acc;
}

// two 16-col tiles sharing the A fragment (32 output columns per wave)
template <int KT>
__device__ __forceinline__ void gemm_x2(const _Float16* A, int lda, const _Float16* Bt, int ldbt,
                                        v8f& c0, v8f& c1) {
#pragma unroll
  for (int t = 0; t < KT; ++t) {
    v16h a = load_a_row(A + t * 32, lda);
    c0 = wmma32(a, load_b_rows(Bt + t * 32, ldbt), c0);
    c1 = wmma32(a, load_b_rows(Bt + (size_t)16 * ldbt + t * 32, ldbt), c1);
  }
}

__device__ __forceinline__ float sigmoidf(float x) { return 1.f / (1.f + __expf(-x)); }

// ---------------- kernels ----------------

// fp32 [K,Nc] row-major -> f16 transposed [Nc,K]
__global__ void k_cvt_t(const float* __restrict__ src, _Float16* __restrict__ dst,
                        int K, int Nc) {
  int i = blockIdx.x * 256 + threadIdx.x;
  if (i < K * Nc) {
    int k = i / Nc, n = i % Nc;
    dst[(size_t)n * K + k] = (_Float16)src[i];
  }
}

// pair init
__global__ void k_init(const int* __restrict__ seq, const float* __restrict__ lw,
                       const float* __restrict__ lb, const float* __restrict__ rw,
                       const float* __restrict__ rb, const float* __restrict__ relw,
                       const float* __restrict__ relb, float* __restrict__ z) {
  int row = blockIdx.x;
  int c = threadIdx.x;
  int i = row >> 8, j = row & 255;
  int si = seq[i], sj = seq[j];
  int d = i - j + 32; d = d < 0 ? 0 : (d > 64 ? 64 : d);
  z[(size_t)row * CH + c] = lw[si * CH + c] + lb[c] + rw[sj * CH + c] + rb[c]
                          + relw[d * CH + c] + relb[c];
}

// LayerNorm over CH=128 channels -> f16; transpose flag reads z^T row
__global__ void k_ln(const float* __restrict__ zin, const float* __restrict__ s,
                     const float* __restrict__ b, _Float16* __restrict__ out, int transpose) {
  __shared__ float red[CH];
  int row = blockIdx.x;
  int src = transpose ? ((row & 255) * NS + (row >> 8)) : row;
  int c = threadIdx.x;
  float x = zin[(size_t)src * CH + c];
  red[c] = x;
  __syncthreads();
  for (int off = CH / 2; off > 0; off >>= 1) {
    if (c < off) red[c] += red[c + off];
    __syncthreads();
  }
  float mu = red[0] * (1.f / CH);
  __syncthreads();
  float d = x - mu;
  red[c] = d * d;
  __syncthreads();
  for (int off = CH / 2; off > 0; off >>= 1) {
    if (c < off) red[c] += red[c + off];
    __syncthreads();
  }
  float var = red[0] * (1.f / CH);
  out[(size_t)row * CH + c] = (_Float16)(d * rsqrtf(var + 1e-5f) * s[c] + b[c]);
}

// tri-mult projection: out = mask2d * (act@pw+pb) * sigmoid(act@gw+gb)
// one A fragment feeds 4 WMMAs (proj lo/hi + gate lo/hi), 32 cols per wave.
// ptrans=1 transposes within the [c] plane (incoming equation).
__global__ void k_proj_gated(const _Float16* __restrict__ act, const _Float16* __restrict__ pwt,
                             const float* __restrict__ pb, const _Float16* __restrict__ gwt,
                             const float* __restrict__ gb, const float* __restrict__ mask,
                             _Float16* __restrict__ out_t, int ptrans) {
  int rt = blockIdx.x, ct = blockIdx.y;   // ct over CH/32
  const _Float16* A  = act + (size_t)rt * 16 * CH;
  const _Float16* P0 = pwt + (size_t)(ct * 32) * CH;
  const _Float16* G0 = gwt + (size_t)(ct * 32) * CH;
  v8f p0 = {}, p1 = {}, g0 = {}, g1 = {};
#pragma unroll
  for (int t = 0; t < 4; ++t) {
    v16h a = load_a_row(A + t * 32, CH);
    p0 = wmma32(a, load_b_rows(P0 + t * 32, CH), p0);
    p1 = wmma32(a, load_b_rows(P0 + (size_t)16 * CH + t * 32, CH), p1);
    g0 = wmma32(a, load_b_rows(G0 + t * 32, CH), g0);
    g1 = wmma32(a, load_b_rows(G0 + (size_t)16 * CH + t * 32, CH), g1);
  }
  int lane = threadIdx.x & 31;
  int col0 = ct * 32 + (lane & 15), col1 = col0 + 16;
  int rb = (lane >> 4) * 8;
  float bp0 = pb[col0], bg0 = gb[col0], bp1 = pb[col1], bg1 = gb[col1];
#pragma unroll
  for (int r = 0; r < 8; ++r) {
    int row = rt * 16 + rb + r;
    float m = mask[row >> 8] * mask[row & 255];
    int dst = ptrans ? ((row & 255) * NS + (row >> 8)) : row;
    out_t[(size_t)col0 * NP + dst] = (_Float16)(m * (p0[r] + bp0) * sigmoidf(g0[r] + bg0));
    out_t[(size_t)col1 * NP + dst] = (_Float16)(m * (p1[r] + bp1) * sigmoidf(g1[r] + bg1));
  }
}

// triangle einsum, per-channel 256x256x256 GEMM with 32x32 register blocking:
// O[i,j,c] = sum_k X_c[i,k] * Y_c[j,k]
__global__ void k_tri_einsum(const _Float16* __restrict__ X, const _Float16* __restrict__ Y,
                             float* __restrict__ outp) {
  int it = blockIdx.x, jt = blockIdx.y, c = blockIdx.z;
  const _Float16* Xc = X + (size_t)c * NP + (size_t)it * 32 * NS;
  const _Float16* Yc = Y + (size_t)c * NP + (size_t)jt * 32 * NS;
  v8f a00 = {}, a01 = {}, a10 = {}, a11 = {};
#pragma unroll
  for (int k0 = 0; k0 < NS; k0 += 32) {
    v16h x0 = load_a_row(Xc + k0, NS);
    v16h x1 = load_a_row(Xc + (size_t)16 * NS + k0, NS);
    v16h y0 = load_b_rows(Yc + k0, NS);
    v16h y1 = load_b_rows(Yc + (size_t)16 * NS + k0, NS);
    a00 = wmma32(x0, y0, a00);
    a01 = wmma32(x0, y1, a01);
    a10 = wmma32(x1, y0, a10);
    a11 = wmma32(x1, y1, a11);
  }
  int lane = threadIdx.x & 31;
  int jc = lane & 15;
  int rbase = (lane >> 4) * 8;
#pragma unroll
  for (int r = 0; r < 8; ++r) {
    int i0 = it * 32 + rbase + r, i1 = i0 + 16;
    int j0 = jt * 32 + jc,        j1 = j0 + 16;
    outp[((size_t)i0 * NS + j0) * CH + c] = a00[r];
    outp[((size_t)i0 * NS + j1) * CH + c] = a01[r];
    outp[((size_t)i1 * NS + j0) * CH + c] = a10[r];
    outp[((size_t)i1 * NS + j1) * CH + c] = a11[r];
  }
}

// z += (act2 @ op_w + op_b) * sigmoid(act @ g_w + g_b), 32 cols per wave
__global__ void k_gate_out(const _Float16* __restrict__ act2, const _Float16* __restrict__ opwt,
                           const float* __restrict__ opb, const _Float16* __restrict__ act,
                           const _Float16* __restrict__ gwt, const float* __restrict__ gb,
                           float* __restrict__ z) {
  int rt = blockIdx.x, ct = blockIdx.y;
  v8f o0 = {}, o1 = {}, g0 = {}, g1 = {};
  gemm_x2<4>(act2 + (size_t)rt * 16 * CH, CH, opwt + (size_t)(ct * 32) * CH, CH, o0, o1);
  gemm_x2<4>(act  + (size_t)rt * 16 * CH, CH, gwt  + (size_t)(ct * 32) * CH, CH, g0, g1);
  int lane = threadIdx.x & 31;
  int col0 = ct * 32 + (lane & 15), col1 = col0 + 16;
  int rb = (lane >> 4) * 8;
  float bo0 = opb[col0], bg0 = gb[col0], bo1 = opb[col1], bg1 = gb[col1];
#pragma unroll
  for (int r = 0; r < 8; ++r) {
    size_t row = (size_t)rt * 16 + rb + r;
    z[row * CH + col0] += (o0[r] + bo0) * sigmoidf(g0[r] + bg0);
    z[row * CH + col1] += (o1[r] + bo1) * sigmoidf(g1[r] + bg1);
  }
}

// attention projections, 32 cols per wave; layout: 0 = [row,CH] (gate),
// 1 = [b,h,n,d] (q,k), 2 = [b,h,d,n] (v transposed for PV B-fragments)
__global__ void k_attn_proj(const _Float16* __restrict__ act, const _Float16* __restrict__ wt,
                            const float* __restrict__ bias, float scale, int do_sig, int layout,
                            _Float16* __restrict__ out) {
  int rt = blockIdx.x, ct = blockIdx.y;
  v8f c0 = {}, c1 = {};
  gemm_x2<4>(act + (size_t)rt * 16 * CH, CH, wt + (size_t)(ct * 32) * CH, CH, c0, c1);
  int lane = threadIdx.x & 31;
  int col0 = ct * 32 + (lane & 15), col1 = col0 + 16;
  int rb = (lane >> 4) * 8;
  float b0 = bias ? bias[col0] : 0.f;
  float b1 = bias ? bias[col1] : 0.f;
#pragma unroll
  for (int r = 0; r < 8; ++r) {
    int row = rt * 16 + rb + r;
    int b = row >> 8, n2 = row & 255;
    float v0 = (c0[r] + b0) * scale;
    float v1 = (c1[r] + b1) * scale;
    if (do_sig) { v0 = sigmoidf(v0); v1 = sigmoidf(v1); }
    int h0 = col0 >> 5, d0 = col0 & 31, h1 = col1 >> 5, d1 = col1 & 31;
    size_t i0, i1;
    if (layout == 1) {
      i0 = (((size_t)b * NH + h0) * NS + n2) * HD + d0;
      i1 = (((size_t)b * NH + h1) * NS + n2) * HD + d1;
    } else if (layout == 2) {
      i0 = (((size_t)b * NH + h0) * HD + d0) * NS + n2;
      i1 = (((size_t)b * NH + h1) * HD + d1) * NS + n2;
    } else {
      i0 = (size_t)row * CH + col0;
      i1 = (size_t)row * CH + col1;
    }
    out[i0] = (_Float16)v0;
    out[i1] = (_Float16)v1;
  }
}

// nonbatched bias: nb[h][q*N+k] = sum_c act[q,k,c] * f2d_w[c,h]
__global__ void k_nb(const _Float16* __restrict__ act, const float* __restrict__ f2d,
                     float* __restrict__ nb) {
  int row = blockIdx.x * 256 + threadIdx.x;
  const _Float16* a = act + (size_t)row * CH;
  float a0 = 0, a1 = 0, a2 = 0, a3 = 0;
  for (int c = 0; c < CH; ++c) {
    float av = (float)a[c];
    a0 += av * f2d[c * 4 + 0];
    a1 += av * f2d[c * 4 + 1];
    a2 += av * f2d[c * 4 + 2];
    a3 += av * f2d[c * 4 + 3];
  }
  nb[0 * (size_t)NP + row] = a0;
  nb[1 * (size_t)NP + row] = a1;
  nb[2 * (size_t)NP + row] = a2;
  nb[3 * (size_t)NP + row] = a3;
}

// attention core per (qtile, h, b): QK^T (+mask bias +nb) -> softmax -> @V -> *sigmoid(g)
__global__ void k_attn(const _Float16* __restrict__ qh, const _Float16* __restrict__ kh,
                       const _Float16* __restrict__ vt, const _Float16* __restrict__ gh,
                       const float* __restrict__ nb, const float* __restrict__ mask,
                       _Float16* __restrict__ og) {
  int qt = blockIdx.x, h = blockIdx.y, b = blockIdx.z;
  int lane = threadIdx.x & 31;
  size_t bh = ((size_t)b * NH + h);
  const _Float16* qb  = qh + bh * NS * HD;   // [n][d]
  const _Float16* kb  = kh + bh * NS * HD;   // [n][d]
  const _Float16* vtb = vt + bh * HD * NS;   // [d][n]

  __shared__ __attribute__((aligned(32))) float    lgts[16 * NS];
  __shared__ __attribute__((aligned(32))) _Float16 wts[16 * NS];
  __shared__ float red[32];

  v16h qa = load_a_row(qb + (size_t)qt * 16 * HD, HD);
  float mb = mask[b];
#pragma unroll
  for (int kt = 0; kt < NS / 16; ++kt) {
    v16h bm = load_b_rows(kb + (size_t)kt * 16 * HD, HD);
    v8f c = {};
    c = wmma32(qa, bm, c);
    int col = kt * 16 + (lane & 15);
    int rb = (lane >> 4) * 8;
    float bias = 1e9f * (mb * mask[col] - 1.f);
    const float* nbq = nb + (size_t)h * NP + (size_t)(qt * 16) * NS;
#pragma unroll
    for (int r = 0; r < 8; ++r)
      lgts[(rb + r) * NS + col] = c[r] + bias + nbq[(size_t)(rb + r) * NS + col];
  }
  __syncthreads();
  {
    int r = lane >> 1, hf = lane & 1;
    float* rowp = lgts + r * NS + hf * 128;
    float mx = -1e30f;
    for (int i = 0; i < 128; ++i) mx = fmaxf(mx, rowp[i]);
    red[lane] = mx;
    __syncthreads();
    mx = fmaxf(red[r * 2], red[r * 2 + 1]);
    __syncthreads();
    float s = 0.f;
    for (int i = 0; i < 128; ++i) { float e = __expf(rowp[i] - mx); rowp[i] = e; s += e; }
    red[lane] = s;
    __syncthreads();
    float inv = 1.f / (red[r * 2] + red[r * 2 + 1]);
    _Float16* wrow = wts + r * NS + hf * 128;
    for (int i = 0; i < 128; ++i) wrow[i] = (_Float16)(rowp[i] * inv);
  }
  __syncthreads();
  v8f o0 = {}, o1 = {};
#pragma unroll
  for (int kt = 0; kt < NS / 32; ++kt) {
    v16h a = load_a_row(wts + kt * 32, NS);
    v16h b0 = load_b_rows(vtb + kt * 32, NS);                      // d 0..15
    v16h b1 = load_b_rows(vtb + (size_t)16 * NS + kt * 32, NS);    // d 16..31
    o0 = wmma32(a, b0, o0);
    o1 = wmma32(a, b1, o1);
  }
  int col = lane & 15, rb2 = (lane >> 4) * 8;
#pragma unroll
  for (int r = 0; r < 8; ++r) {
    size_t rowi = (size_t)b * NS + qt * 16 + rb2 + r;
    int c0 = h * HD + col, c1 = c0 + 16;
    og[rowi * CH + c0] = (_Float16)(o0[r] * (float)gh[rowi * CH + c0]);
    og[rowi * CH + c1] = (_Float16)(o1[r] * (float)gh[rowi * CH + c1]);
  }
}

// attention output projection + residual, 32 cols per wave
__global__ void k_attn_out(const _Float16* __restrict__ og, const _Float16* __restrict__ owt,
                           const float* __restrict__ ob, float* __restrict__ z, int transpose) {
  int rt = blockIdx.x, ct = blockIdx.y;
  v8f c0 = {}, c1 = {};
  gemm_x2<4>(og + (size_t)rt * 16 * CH, CH, owt + (size_t)(ct * 32) * CH, CH, c0, c1);
  int lane = threadIdx.x & 31;
  int col0 = ct * 32 + (lane & 15), col1 = col0 + 16;
  int rb = (lane >> 4) * 8;
  float b0 = ob[col0], b1 = ob[col1];
#pragma unroll
  for (int r = 0; r < 8; ++r) {
    int row = rt * 16 + rb + r;
    int dst = transpose ? ((row & 255) * NS + (row >> 8)) : row;
    z[(size_t)dst * CH + col0] += c0[r] + b0;
    z[(size_t)dst * CH + col1] += c1[r] + b1;
  }
}

// fused transition: z += relu(act@w1+b1)@w2 + b2, hidden staged in LDS (f16)
__global__ void k_transition(const _Float16* __restrict__ act, const _Float16* __restrict__ w1t,
                             const float* __restrict__ b1, const _Float16* __restrict__ w2t,
                             const float* __restrict__ b2, float* __restrict__ z) {
  __shared__ __attribute__((aligned(32))) _Float16 hid[16 * 512];
  int wv = threadIdx.x >> 5;
  int lane = threadIdx.x & 31;
  const _Float16* A = act + (size_t)blockIdx.x * 16 * CH;
  for (int ct = wv; ct < 16; ct += 8) {  // 16 x2-tiles cover 512 cols
    v8f h0 = {}, h1 = {};
    gemm_x2<4>(A, CH, w1t + (size_t)(ct * 32) * CH, CH, h0, h1);
    int col0 = ct * 32 + (lane & 15), col1 = col0 + 16;
    int rb = (lane >> 4) * 8;
    float bb0 = b1[col0], bb1 = b1[col1];
#pragma unroll
    for (int r = 0; r < 8; ++r) {
      hid[(rb + r) * 512 + col0] = (_Float16)fmaxf(h0[r] + bb0, 0.f);
      hid[(rb + r) * 512 + col1] = (_Float16)fmaxf(h1[r] + bb1, 0.f);
    }
  }
  __syncthreads();
  {
    int ct = wv;  // 8 waves -> 8 col tiles of 16 (CH=128)
    v8f acc = gemm_tile_t<16>((const _Float16*)hid, 512, w2t + (size_t)(ct * 16) * 512, 512);
    int col = ct * 16 + (lane & 15);
    int rb = (lane >> 4) * 8;
    float bb = b2[col];
#pragma unroll
    for (int r = 0; r < 8; ++r) {
      size_t row = (size_t)blockIdx.x * 16 + rb + r;
      z[row * CH + col] += acc[r] + bb;
    }
  }
}

// distogram head: out[i,j,:] = sigmoid(z[i,j]@pw+pb) + sigmoid(z[j,i]@pw+pb)
__global__ void k_post(const float* __restrict__ z, const float* __restrict__ pw,
                       const float* __restrict__ pb, float* __restrict__ out) {
  int idx = blockIdx.x * 256 + threadIdx.x;
  int i = idx >> 8, j = idx & 255;
  const float* zij = z + (size_t)idx * CH;
  const float* zji = z + ((size_t)(j * NS + i)) * CH;
  float a0 = pb[0], a1 = pb[1], c0 = pb[0], c1 = pb[1];
  for (int c = 0; c < CH; ++c) {
    float w0 = pw[c * 2], w1 = pw[c * 2 + 1];
    a0 += zij[c] * w0; a1 += zij[c] * w1;
    c0 += zji[c] * w0; c1 += zji[c] * w1;
  }
  out[(size_t)idx * 2 + 0] = sigmoidf(a0) + sigmoidf(c0);
  out[(size_t)idx * 2 + 1] = sigmoidf(a1) + sigmoidf(c1);
}

// ---------------- host launcher ----------------
// Input index map (setup_inputs insertion order):
// 0 left_w 1 left_b 2 right_w 3 right_b 4 relpos_w 5 relpos_b
// tmo base 6 / tmi base 22: +0 ln_s +1 ln_b +2 lp_w +3 lp_b +4 rp_w +5 rp_b +6 lg_w +7 lg_b
//                           +8 rg_w +9 rg_b +10 cln_s +11 cln_b +12 op_w +13 op_b +14 g_w +15 g_b
// tas base 38 / tae base 48: +0 ln_s +1 ln_b +2 q_w +3 k_w +4 v_w +5 f2d_w +6 g_w +7 g_b +8 o_w +9 o_b
// tr base 58: ln_s ln_b w1 b1 w2 b2 ; 64 post_w 65 post_b 66 seq 67 mask
extern "C" void kernel_launch(void* const* d_in, const int* in_sizes, int n_in,
                              void* d_out, int out_size, void* d_ws, size_t ws_size,
                              hipStream_t stream) {
  (void)in_sizes; (void)n_in; (void)out_size; (void)ws_size;
  auto F = [&](int i) { return (const float*)d_in[i]; };
  const int* seq = (const int*)d_in[66];
  const float* mask = F(67);

  const size_t MB = 1ull << 20;
  char* ws = (char*)d_ws;
  float*    z    = (float*)(ws);                 // 32 MB fp32 master pair rep
  _Float16* act  = (_Float16*)(ws + 32 * MB);    // 16 MB f16 LN(z)
  _Float16* act2 = (_Float16*)(ws + 48 * MB);    // 16 MB f16 LN(tmp)
  char*     wb   = ws + 64 * MB;                 // <2 MB f16 transposed-weight arena
  char*     un   = ws + 66 * MB;                 // phase-union scratch (81 MB)
  _Float16* Lt   = (_Float16*)(un);              // [c][N*N] f16 (16 MB)
  _Float16* Rt   = (_Float16*)(un + 16 * MB);    // 16 MB
  float*    tmp  = (float*)(un + 32 * MB);       // 32 MB f32 einsum out
  _Float16* qh   = (_Float16*)(un);              // attention phase reuses union
  _Float16* kh   = (_Float16*)(un + 16 * MB);
  _Float16* vh   = (_Float16*)(un + 32 * MB);
  _Float16* gh   = (_Float16*)(un + 48 * MB);
  _Float16* og   = (_Float16*)(un + 64 * MB);
  float*    nbuf = (float*)(un + 80 * MB);       // 1 MB

  size_t woff = 0;
  auto cvtT = [&](int idx, int K, int Nc) -> _Float16* {
    _Float16* dst = (_Float16*)(wb + woff);
    int n = K * Nc;
    woff += (size_t)n * 2;
    k_cvt_t<<<dim3((n + 255) / 256), dim3(256), 0, stream>>>(F(idx), dst, K, Nc);
    return dst;
  };

  _Float16 *tmo_lp = cvtT(6 + 2, CH, CH),  *tmo_rp = cvtT(6 + 4, CH, CH);
  _Float16 *tmo_lg = cvtT(6 + 6, CH, CH),  *tmo_rg = cvtT(6 + 8, CH, CH);
  _Float16 *tmo_op = cvtT(6 + 12, CH, CH), *tmo_g  = cvtT(6 + 14, CH, CH);
  _Float16 *tmi_lp = cvtT(22 + 2, CH, CH), *tmi_rp = cvtT(22 + 4, CH, CH);
  _Float16 *tmi_lg = cvtT(22 + 6, CH, CH), *tmi_rg = cvtT(22 + 8, CH, CH);
  _Float16 *tmi_op = cvtT(22 + 12, CH, CH),*tmi_g  = cvtT(22 + 14, CH, CH);
  _Float16 *tas_q = cvtT(38 + 2, CH, CH), *tas_k = cvtT(38 + 3, CH, CH), *tas_v = cvtT(38 + 4, CH, CH);
  _Float16 *tas_g = cvtT(38 + 6, CH, CH), *tas_o = cvtT(38 + 8, CH, CH);
  _Float16 *tae_q = cvtT(48 + 2, CH, CH), *tae_k = cvtT(48 + 3, CH, CH), *tae_v = cvtT(48 + 4, CH, CH);
  _Float16 *tae_g = cvtT(48 + 6, CH, CH), *tae_o = cvtT(48 + 8, CH, CH);
  _Float16 *tr_w1 = cvtT(58 + 2, CH, 512), *tr_w2 = cvtT(58 + 4, 512, CH);

  k_init<<<dim3(NP), dim3(CH), 0, stream>>>(seq, F(0), F(1), F(2), F(3), F(4), F(5), z);

  auto tri_phase = [&](int base, _Float16* lp, _Float16* rp, _Float16* lg_, _Float16* rg,
                       _Float16* op, _Float16* gw, int incoming) {
    k_ln<<<dim3(NP), dim3(CH), 0, stream>>>(z, F(base + 0), F(base + 1), act, 0);
    k_proj_gated<<<dim3(NP / 16, CH / 32), dim3(32), 0, stream>>>(
        act, lp, F(base + 3), lg_, F(base + 7), mask, Lt, incoming);
    k_proj_gated<<<dim3(NP / 16, CH / 32), dim3(32), 0, stream>>>(
        act, rp, F(base + 5), rg, F(base + 9), mask, Rt, incoming);
    if (!incoming)
      k_tri_einsum<<<dim3(NS / 32, NS / 32, CH), dim3(32), 0, stream>>>(Lt, Rt, tmp);
    else
      k_tri_einsum<<<dim3(NS / 32, NS / 32, CH), dim3(32), 0, stream>>>(Rt, Lt, tmp);
    k_ln<<<dim3(NP), dim3(CH), 0, stream>>>(tmp, F(base + 10), F(base + 11), act2, 0);
    k_gate_out<<<dim3(NP / 16, CH / 32), dim3(32), 0, stream>>>(
        act2, op, F(base + 13), act, gw, F(base + 15), z);
  };
  tri_phase(6,  tmo_lp, tmo_rp, tmo_lg, tmo_rg, tmo_op, tmo_g, 0);
  tri_phase(22, tmi_lp, tmi_rp, tmi_lg, tmi_rg, tmi_op, tmi_g, 1);

  auto attn_phase = [&](int base, _Float16* qw, _Float16* kw, _Float16* vw,
                        _Float16* gw, _Float16* ow, int percol) {
    k_ln<<<dim3(NP), dim3(CH), 0, stream>>>(z, F(base + 0), F(base + 1), act, percol);
    k_attn_proj<<<dim3(NP / 16, CH / 32), dim3(32), 0, stream>>>(
        act, qw, nullptr, 0.17677669529f, 0, 1, qh);
    k_attn_proj<<<dim3(NP / 16, CH / 32), dim3(32), 0, stream>>>(act, kw, nullptr, 1.f, 0, 1, kh);
    k_attn_proj<<<dim3(NP / 16, CH / 32), dim3(32), 0, stream>>>(act, vw, nullptr, 1.f, 0, 2, vh);
    k_attn_proj<<<dim3(NP / 16, CH / 32), dim3(32), 0, stream>>>(
        act, gw, F(base + 7), 1.f, 1, 0, gh);
    k_nb<<<dim3(NP / 256), dim3(256), 0, stream>>>(act, F(base + 5), nbuf);
    k_attn<<<dim3(NS / 16, NH, NS), dim3(32), 0, stream>>>(qh, kh, vh, gh, nbuf, mask, og);
    k_attn_out<<<dim3(NP / 16, CH / 32), dim3(32), 0, stream>>>(og, ow, F(base + 9), z, percol);
  };
  attn_phase(38, tas_q, tas_k, tas_v, tas_g, tas_o, 0);
  attn_phase(48, tae_q, tae_k, tae_v, tae_g, tae_o, 1);

  k_ln<<<dim3(NP), dim3(CH), 0, stream>>>(z, F(58), F(59), act, 0);
  k_transition<<<dim3(NP / 16), dim3(256), 0, stream>>>(act, tr_w1, F(61), tr_w2, F(63), z);

  k_post<<<dim3(NP / 256), dim3(256), 0, stream>>>(z, F(64), F(65), (float*)d_out);
}
